// MultiHeadAttention_3453153706654
// MI455X (gfx1250) — matrix operand order repaired and checked by reading, not compiled, not run on hardware
//
#include <hip/hip_runtime.h>

// ---------------------------------------------------------------------------
// MHA forward for MI455X (gfx1250, wave32, WMMA).
// Pre-convert fp32 operands to bf16 once (plus proj_kernel transpose), then:
// QKV proj -> logits -> row softmax -> P@V -> out proj, all bf16 WMMA with
// f32 accumulation. Double-buffered LDS; bf16 operands staged directly with
// GLOBAL_LOAD_ASYNC_TO_LDS_B64 (ASYNCcnt) — no VALU in those stage paths.
// ---------------------------------------------------------------------------

typedef __attribute__((ext_vector_type(16))) __bf16 v16bf;
typedef __attribute__((ext_vector_type(8)))  float  v8f;

constexpr int Bb = 2, Tt = 2048, Hh = 16, Dd = 64, Oo = 1024, Ee = Hh * Dd;

constexpr int BM = 128, BK = 32;
constexpr int APAD = BK + 8;  // 40 bf16 = 80B row stride (16B aligned, conflict-free)
constexpr int BPAD = BK + 8;

// ---- fp32 -> bf16, round-to-nearest, pair packed with one v_perm_b32 ----
__device__ __forceinline__ unsigned pack2(float a, float b) {
  union { float f; unsigned u; } x, y;
  x.f = a; y.f = b;
  unsigned au = x.u + 0x8000u, bu = y.u + 0x8000u;
#if __has_builtin(__builtin_amdgcn_perm)
  return __builtin_amdgcn_perm(bu, au, 0x07060302u);  // { bu[31:16], au[31:16] }
#else
  return (bu & 0xFFFF0000u) | (au >> 16);
#endif
}
__device__ __forceinline__ unsigned short f2bf1(float f) {
  union { float f; unsigned u; } x; x.f = f;
  return (unsigned short)((x.u + 0x8000u) >> 16);
}

union FragAB { uint4 q[2]; v16bf v; };

template <typename T> struct LdVec;
template <> struct LdVec<float>          { using V = float4; };
template <> struct LdVec<unsigned short> { using V = uint2;  };

__device__ __forceinline__ uint2 to_bf4(float4 d) {
  return make_uint2(pack2(d.x, d.y), pack2(d.z, d.w));
}
__device__ __forceinline__ uint2 to_bf4(uint2 d) { return d; }

// Direct memory -> LDS async copy of 8 bytes (4 bf16), GVS addressing.
__device__ __forceinline__ void async_b64(const void* base, unsigned byte_off,
                                          void* lds_ptr) {
  unsigned lds_addr = (unsigned)(unsigned long long)lds_ptr;  // low 32b = LDS offset
  asm volatile("global_load_async_to_lds_b64 %0, %1, %2"
               :
               : "v"(lds_addr), "v"(byte_off),
                 "s"((unsigned long long)base)
               : "memory");
}
__device__ __forceinline__ void wait_async0() {
  asm volatile("s_wait_asynccnt 0x0" ::: "memory");
}

// C[m,n] = alpha * sum_k A[m,k] * B(k,n) (+ bias[n])
// B_KROW=true : B(k,n) = Bg[n*ldb + k]
// B_KROW=false: B(k,n) = Bg[k*ldb + n]
// DO_SCALE=false lets the compiler drop the alpha multiply entirely.
template <typename AT, typename BT, bool B_KROW, typename CT, bool BIAS,
          bool DO_SCALE, int BNT>
__global__ __launch_bounds__(256) void wmma_gemm(
    const AT* __restrict__ Ag, int lda,
    const BT* __restrict__ Bg, int ldb,
    CT* __restrict__ Cg, int ldc,
    int Kdim, float alpha, const float* __restrict__ bias,
    int Hdiv,
    long long sAb, long long sAh,
    long long sBb, long long sBh,
    long long sCb, long long sCh)
{
  constexpr int NF = BNT / 32;   // B fragments per wave (N direction)
  constexpr int NB = BNT / 32;   // B stage iterations
  constexpr int NQ = BNT / 4;    // quads per k-row in non-KROW stage
  constexpr bool ASYNC_A = (sizeof(AT) == 2);              // bf16, contiguous copy
  constexpr bool ASYNC_B = (sizeof(BT) == 2) && B_KROW;    // bf16, contiguous copy

  __shared__ __align__(16) unsigned short As[2][BM * APAD];
  __shared__ __align__(16) unsigned short Bs[2][BNT * BPAD];

  const int z  = blockIdx.z;
  const int zb = z / Hdiv, zh = z % Hdiv;
  const AT* A  = Ag + zb * sAb + zh * sAh;
  const BT* Bm = Bg + zb * sBb + zh * sBh;
  CT*       C  = Cg + zb * sCb + zh * sCh;

  const int m0 = blockIdx.y * BM;
  const int n0 = blockIdx.x * BNT;

  const int tid  = threadIdx.x;
  const int lane = tid & 31;
  const int wave = tid >> 5;
  const int wm   = wave & 3;   // 4 waves along M (32 rows each)
  const int wn   = wave >> 2;  // 2 waves along N (BNT/2 cols each)

  using AV = typename LdVec<AT>::V;
  using BV = typename LdVec<BT>::V;
  AV areg[4];
  BV breg[NB];

  auto stageA_async = [&](int buf, int kt) {
    #pragma unroll
    for (int i = 0; i < 4; ++i) {
      int f   = i * 256 + tid;
      int row = f >> 3;
      int kg  = (f & 7) * 4;
      unsigned off = (unsigned)(((long long)(m0 + row) * lda + kt + kg) * 2);
      async_b64(A, off, &As[buf][row * APAD + kg]);
    }
  };
  auto loadA = [&](int kt) {
    #pragma unroll
    for (int i = 0; i < 4; ++i) {
      int f   = i * 256 + tid;
      int row = f >> 3;
      int kg  = (f & 7) * 4;
      areg[i] = *(const AV*)(A + (long long)(m0 + row) * lda + kt + kg);
    }
  };
  auto storeA = [&](int buf) {
    #pragma unroll
    for (int i = 0; i < 4; ++i) {
      int f   = i * 256 + tid;
      int row = f >> 3;
      int kg  = (f & 7) * 4;
      *(uint2*)&As[buf][row * APAD + kg] = to_bf4(areg[i]);
    }
  };
  auto stageB_async = [&](int buf, int kt) {
    #pragma unroll
    for (int i = 0; i < NB; ++i) {
      int f  = i * 256 + tid;
      int n  = f >> 3;
      int kg = (f & 7) * 4;
      unsigned off = (unsigned)(((long long)(n0 + n) * ldb + kt + kg) * 2);
      async_b64(Bm, off, &Bs[buf][n * BPAD + kg]);
    }
  };
  auto loadB = [&](int kt) {
    #pragma unroll
    for (int i = 0; i < NB; ++i) {
      int f = i * 256 + tid;
      long long src;
      if constexpr (B_KROW) {
        int n  = f >> 3;
        int kg = (f & 7) * 4;
        src = (long long)(n0 + n) * ldb + kt + kg;
      } else {
        int k  = f / NQ;
        int ng = (f % NQ) * 4;
        src = (long long)(kt + k) * ldb + n0 + ng;
      }
      breg[i] = *(const BV*)(Bm + src);
    }
  };
  auto storeB = [&](int buf) {
    #pragma unroll
    for (int i = 0; i < NB; ++i) {
      int f = i * 256 + tid;
      if constexpr (B_KROW) {
        int n  = f >> 3;
        int kg = (f & 7) * 4;
        *(uint2*)&Bs[buf][n * BPAD + kg] = to_bf4(breg[i]);
      } else {
        int k  = f / NQ;
        int ng = (f % NQ) * 4;
        if constexpr (sizeof(BT) == 4) {
          float4 d = breg[i];
          Bs[buf][(ng + 0) * BPAD + k] = f2bf1(d.x);
          Bs[buf][(ng + 1) * BPAD + k] = f2bf1(d.y);
          Bs[buf][(ng + 2) * BPAD + k] = f2bf1(d.z);
          Bs[buf][(ng + 3) * BPAD + k] = f2bf1(d.w);
        } else {
          uint2 d = breg[i];
          Bs[buf][(ng + 0) * BPAD + k] = (unsigned short)(d.x & 0xFFFFu);
          Bs[buf][(ng + 1) * BPAD + k] = (unsigned short)(d.x >> 16);
          Bs[buf][(ng + 2) * BPAD + k] = (unsigned short)(d.y & 0xFFFFu);
          Bs[buf][(ng + 3) * BPAD + k] = (unsigned short)(d.y >> 16);
        }
      }
    }
  };

  v8f acc[2][NF] = {};

  // ---- prologue: stage first tile ----
  if constexpr (ASYNC_A) stageA_async(0, 0); else { loadA(0); storeA(0); }
  if constexpr (ASYNC_B) stageB_async(0, 0); else { loadB(0); storeB(0); }
  if constexpr (ASYNC_A || ASYNC_B) wait_async0();
  __syncthreads();

  int buf = 0;
  for (int kt = 0; kt < Kdim; kt += BK) {
    const bool more = (kt + BK) < Kdim;
    if (more) {
      // buf^1 was fully consumed before the previous barrier: safe to fill now.
      if constexpr (ASYNC_A) stageA_async(buf ^ 1, kt + BK); else loadA(kt + BK);
      if constexpr (ASYNC_B) stageB_async(buf ^ 1, kt + BK); else loadB(kt + BK);
    }

    // ---- fragments per ISA wave32 16-bit layouts ----
    FragAB af[2], bfr[NF];
    const int mrow = lane & 15;
    const int half = lane >> 4;
    #pragma unroll
    for (int i = 0; i < 2; ++i) {
      int row = wm * 32 + i * 16 + mrow;
      af[i].q[0] = *(const uint4*)&As[buf][row * APAD + half * 8];
      af[i].q[1] = *(const uint4*)&As[buf][row * APAD + 16 + half * 8];
    }
    #pragma unroll
    for (int j = 0; j < NF; ++j) {
      int n = wn * (BNT / 2) + j * 16 + mrow;
      bfr[j].q[0] = *(const uint4*)&Bs[buf][n * BPAD + half * 8];
      bfr[j].q[1] = *(const uint4*)&Bs[buf][n * BPAD + 16 + half * 8];
    }
    #pragma unroll
    for (int i = 0; i < 2; ++i)
      #pragma unroll
      for (int j = 0; j < NF; ++j)
        acc[i][j] = __builtin_amdgcn_wmma_f32_16x16x32_bf16(
            false, af[i].v, false, bfr[j].v, (short)0, acc[i][j], false, false);

    if (more) {
      if constexpr (!ASYNC_A) storeA(buf ^ 1);
      if constexpr (!ASYNC_B) storeB(buf ^ 1);
      if constexpr (ASYNC_A || ASYNC_B) wait_async0();
    }
    __syncthreads();
    buf ^= 1;
  }

  // ---- writeback: VGPR r -> M = r + 8*(lane>=16), N = lane%16 ----
  const int ln  = lane & 15;
  const int mh8 = (lane >> 4) * 8;
  #pragma unroll
  for (int i = 0; i < 2; ++i) {
    #pragma unroll
    for (int j = 0; j < NF; ++j) {
      int col = n0 + wn * (BNT / 2) + j * 16 + ln;
      float bv = 0.0f;
      if constexpr (BIAS) bv = bias[col];
      #pragma unroll
      for (int r = 0; r < 8; ++r) {
        int row = m0 + wm * 32 + i * 16 + mh8 + r;
        float v = acc[i][j][r];
        if constexpr (DO_SCALE) v *= alpha;
        if constexpr (BIAS) v += bv;
        if constexpr (sizeof(CT) == 4)
          C[(long long)row * ldc + col] = v;
        else
          C[(long long)row * ldc + col] = f2bf1(v);
      }
    }
  }
}

// Elementwise fp32 -> bf16 (4 elements/thread).
__global__ __launch_bounds__(256) void cvt_bf16(const float* __restrict__ in,
                                                unsigned short* __restrict__ out,
                                                long long n4) {
  long long i = (long long)blockIdx.x * 256 + threadIdx.x;
  if (i < n4) {
    float4 d = ((const float4*)in)[i];
    ((uint2*)out)[i] = to_bf4(d);
  }
}

// Transpose [K,O] fp32 -> [O,K] bf16 (32x32 LDS tiles).
__global__ __launch_bounds__(256) void transpose_cvt(const float* __restrict__ in,
                                                     unsigned short* __restrict__ out,
                                                     int K, int O) {
  __shared__ float t[32][33];
  const int o0 = blockIdx.x * 32, k0 = blockIdx.y * 32;
  const int tx = threadIdx.x, ty = threadIdx.y;
  #pragma unroll
  for (int r = ty; r < 32; r += 8)
    t[r][tx] = in[(long long)(k0 + r) * O + o0 + tx];
  __syncthreads();
  #pragma unroll
  for (int r = ty; r < 32; r += 8)
    out[(long long)(o0 + r) * K + k0 + tx] = f2bf1(t[tx][r]);
}

// In-place softmax over rows of length 2048 (one 256-thread block per row).
__global__ __launch_bounds__(256) void softmax_rows(float* __restrict__ attn) {
  __shared__ float red[256];
  float* p = attn + (long long)blockIdx.x * Tt;
  const int tid = threadIdx.x;
  float v[8];
  float mx = -3.4e38f;
  #pragma unroll
  for (int i = 0; i < 8; ++i) { v[i] = p[tid + i * 256]; mx = fmaxf(mx, v[i]); }
  red[tid] = mx; __syncthreads();
  for (int s = 128; s > 0; s >>= 1) {
    if (tid < s) red[tid] = fmaxf(red[tid], red[tid + s]);
    __syncthreads();
  }
  mx = red[0]; __syncthreads();
  float sum = 0.f;
  #pragma unroll
  for (int i = 0; i < 8; ++i) { v[i] = __expf(v[i] - mx); sum += v[i]; }
  red[tid] = sum; __syncthreads();
  for (int s = 128; s > 0; s >>= 1) {
    if (tid < s) red[tid] += red[tid + s];
    __syncthreads();
  }
  float inv = 1.0f / red[0];
  #pragma unroll
  for (int i = 0; i < 8; ++i) p[tid + i * 256] = v[i] * inv;
}

extern "C" void kernel_launch(void* const* d_in, const int* in_sizes, int n_in,
                              void* d_out, int out_size, void* d_ws, size_t ws_size,
                              hipStream_t stream) {
  const float* query = (const float*)d_in[0];
  const float* key_  = (const float*)d_in[1];
  const float* value = (const float*)d_in[2];
  const float* Wq    = (const float*)d_in[3];
  const float* Wk    = (const float*)d_in[4];
  const float* Wv    = (const float*)d_in[5];
  const float* Wp    = (const float*)d_in[6];  // [H,D,O] == [E,O] row-major
  const float* bias  = (const float*)d_in[7];

  float* out  = (float*)d_out;                    // [B,T,O]
  float* attn = out + (long long)Bb * Tt * Oo;    // [B,H,T,T]

  // bf16 workspace (~64 MB total)
  const long long NTE = (long long)Bb * Tt * Ee;  // 4M
  const long long NEE = (long long)Ee * Ee;       // 1M
  unsigned short* qb  = (unsigned short*)d_ws;    // projected q (scaled)
  unsigned short* kb  = qb  + NTE;
  unsigned short* vb  = kb  + NTE;
  unsigned short* mhb = vb  + NTE;                // attn @ V
  unsigned short* xq  = mhb + NTE;                // converted inputs
  unsigned short* xk  = xq  + NTE;
  unsigned short* xv  = xk  + NTE;
  unsigned short* wqb = xv  + NTE;                // converted weights
  unsigned short* wkb = wqb + NEE;
  unsigned short* wvb = wkb + NEE;
  unsigned short* wtb = wvb + NEE;                // proj_kernel^T [O,E]

  dim3 blk(256);

  // 0) one-time bf16 conversion passes (bandwidth-trivial: ~100 MB total)
  cvt_bf16<<<dim3((unsigned)(NTE / 4 / 256)), blk, 0, stream>>>(query, xq, NTE / 4);
  cvt_bf16<<<dim3((unsigned)(NTE / 4 / 256)), blk, 0, stream>>>(key_,  xk, NTE / 4);
  cvt_bf16<<<dim3((unsigned)(NTE / 4 / 256)), blk, 0, stream>>>(value, xv, NTE / 4);
  cvt_bf16<<<dim3((unsigned)(NEE / 4 / 256)), blk, 0, stream>>>(Wq, wqb, NEE / 4);
  cvt_bf16<<<dim3((unsigned)(NEE / 4 / 256)), blk, 0, stream>>>(Wk, wkb, NEE / 4);
  cvt_bf16<<<dim3((unsigned)(NEE / 4 / 256)), blk, 0, stream>>>(Wv, wvb, NEE / 4);
  transpose_cvt<<<dim3(Oo / 32, Ee / 32), dim3(32, 8), 0, stream>>>(Wp, wtb, Ee, Oo);

  // 1) QKV projections (fully async-staged): [B*T,E] x W^T -> bf16
  {
    dim3 g(Ee / 128, (Bb * Tt) / BM, 1);
    wmma_gemm<unsigned short, unsigned short, true, unsigned short, false, true, 128>
        <<<g, blk, 0, stream>>>(xq, Ee, wqb, Ee, qb, Ee, Ee, 0.125f, nullptr,
                                1, 0, 0, 0, 0, 0, 0);
    wmma_gemm<unsigned short, unsigned short, true, unsigned short, false, false, 128>
        <<<g, blk, 0, stream>>>(xk, Ee, wkb, Ee, kb, Ee, Ee, 1.0f, nullptr,
                                1, 0, 0, 0, 0, 0, 0);
    wmma_gemm<unsigned short, unsigned short, true, unsigned short, false, false, 128>
        <<<g, blk, 0, stream>>>(xv, Ee, wvb, Ee, vb, Ee, Ee, 1.0f, nullptr,
                                1, 0, 0, 0, 0, 0, 0);
  }
  // 2) logits[b,h,n,m] = q[b,n,h,:] . k[b,m,h,:]  (fully async-staged)
  {
    dim3 g(Tt / 128, Tt / BM, Bb * Hh);
    wmma_gemm<unsigned short, unsigned short, true, float, false, false, 128>
        <<<g, blk, 0, stream>>>(qb, Ee, kb, Ee, attn, Tt, Dd, 1.0f, nullptr, Hh,
                                (long long)Tt * Ee, (long long)Dd,
                                (long long)Tt * Ee, (long long)Dd,
                                (long long)Hh * Tt * Tt, (long long)Tt * Tt);
  }
  // 3) softmax over last axis, in place
  softmax_rows<<<dim3(Bb * Hh * Tt), blk, 0, stream>>>(attn);
  // 4) mh[b,n,h,:] = P[b,h,n,:] @ v[b,:,h,:]  (probs fp32 -> bf16 in stage,
  //    each prob element staged exactly once so no pre-pass is cheaper)
  {
    dim3 g(Dd / 64, Tt / BM, Bb * Hh);
    wmma_gemm<float, unsigned short, false, unsigned short, false, false, 64>
        <<<g, blk, 0, stream>>>(attn, Tt, vb, Ee, mhb, Ee, Tt, 1.0f, nullptr, Hh,
                                (long long)Hh * Tt * Tt, (long long)Tt * Tt,
                                (long long)Tt * Ee, (long long)Dd,
                                (long long)Tt * Ee, (long long)Dd);
  }
  // 5) out = mh[B*T,E] @ projT[O,E]^T + bias  (fully async-staged)
  {
    dim3 g(Oo / 128, (Bb * Tt) / BM, 1);
    wmma_gemm<unsigned short, unsigned short, true, float, true, false, 128>
        <<<g, blk, 0, stream>>>(mhb, Ee, wtb, Ee, out, Oo, Ee, 1.0f, bias,
                                1, 0, 0, 0, 0, 0, 0);
  }
}
